// SelfAttention_5050881540762
// MI455X (gfx1250) — matrix operand order repaired and checked by reading, not compiled, not run on hardware
//
#include <hip/hip_runtime.h>
#include <hip/hip_fp16.h>

typedef __attribute__((ext_vector_type(16))) _Float16 v16h;
typedef __attribute__((ext_vector_type(8)))  _Float16 v8h;
typedef __attribute__((ext_vector_type(8)))  float    v8f;

#define WMMA16x16x32(a, b, c) \
  __builtin_amdgcn_wmma_f32_16x16x32_f16(false, (a), false, (b), (short)0, (c), false, false)

union FragU { v16h v; uint4 q[2]; };
union PackU { v8h  h; uint4 q;    };

// A-operand (16x32 f16, M x K): lane holds row m = lane%16.
// lanes 0-15: K = k0+0..7 and k0+16..23 ; lanes 16-31: +8 shift.
static __device__ __forceinline__ v16h load_a16(const _Float16* rowp, int lane) {
  const int kk = (lane & 16) ? 8 : 0;
  FragU f;
  f.q[0] = *reinterpret_cast<const uint4*>(rowp + kk);
  f.q[1] = *reinterpret_cast<const uint4*>(rowp + kk + 16);
  return f.v;
}
// B-operand (32x16 f16, K x N): lane holds column n = lane%16.
// lanes 0-15: K = k0+0..7 and k0+16..23 ; lanes 16-31: +8 shift.
static __device__ __forceinline__ v16h load_b16(const _Float16* rowp, int lane) {
  const int kk = (lane & 16) ? 8 : 0;
  FragU f;
  f.q[0] = *reinterpret_cast<const uint4*>(rowp + kk);
  f.q[1] = *reinterpret_cast<const uint4*>(rowp + kk + 16);
  return f.v;
}

// ---------------------------------------------------------------- prep kernels
__global__ void k_cvt_x(const float* __restrict__ x, _Float16* __restrict__ xh, int n4) {
  const int i = blockIdx.x * blockDim.x + threadIdx.x;
  if (i >= n4) return;
  const float4 v = reinterpret_cast<const float4*>(x)[i];
  union { _Float16 h[4]; uint2 u; } p;
  p.h[0] = (_Float16)v.x; p.h[1] = (_Float16)v.y;
  p.h[2] = (_Float16)v.z; p.h[3] = (_Float16)v.w;
  reinterpret_cast<uint2*>(xh)[i] = p.u;
}

__global__ void k_wt(const float* __restrict__ W0, const float* __restrict__ W1,
                     const float* __restrict__ W2, _Float16* __restrict__ Wt) {
  const int d = blockIdx.x * blockDim.x + threadIdx.x;   // 0..767
  const int u = blockIdx.y;                               // 0..767
  const int z = blockIdx.z;                               // 0..2
  const float* W = (z == 0) ? W0 : ((z == 1) ? W1 : W2);
  Wt[((size_t)z * 768 + u) * 768 + d] = (_Float16)W[(size_t)d * 768 + u];
}

// ---------------------------------------------------------------- QKV GEMM
// grid (128, 12, 3), block 256. z selects Wq/Wk/Wv. Q,K row-major [8192,768];
// V stored transposed per batch: VT[b][u][s].
__global__ __launch_bounds__(256) void k_qkv(const _Float16* __restrict__ xh,
                                             const _Float16* __restrict__ Wt,
                                             _Float16* __restrict__ Qh,
                                             _Float16* __restrict__ Kh,
                                             _Float16* __restrict__ VT) {
  const int tid  = threadIdx.x;
  const int lane = tid & 31;
  const int w    = tid >> 5;                   // 8 waves
  const int z    = blockIdx.z;
  const _Float16* Wz = Wt + (size_t)z * 768 * 768;

  const int m_base = blockIdx.x * 64 + (w >> 1) * 16;
  const int n_base = blockIdx.y * 64 + (w & 1) * 32;
  const int arow   = m_base + (lane & 15);
  const int brow   = n_base + (lane & 15);

  v8f acc0 = {}, acc1 = {};
  for (int k0 = 0; k0 < 768; k0 += 32) {
    const v16h a  = load_a16(xh + (size_t)arow * 768 + k0, lane);
    const v16h b0 = load_b16(Wz + (size_t)brow * 768 + k0, lane);
    const v16h b1 = load_b16(Wz + (size_t)(brow + 16) * 768 + k0, lane);
    acc0 = WMMA16x16x32(a, b0, acc0);
    acc1 = WMMA16x16x32(a, b1, acc1);
  }

  const int roff = (lane & 16) ? 8 : 0;        // C/D layout: M = r + 8*(lane>=16)
  if (z < 2) {
    _Float16* out = (z == 0) ? Qh : Kh;
    #pragma unroll
    for (int r = 0; r < 8; ++r) {
      const int row = m_base + roff + r;
      out[(size_t)row * 768 + n_base + (lane & 15)]      = (_Float16)acc0[r];
      out[(size_t)row * 768 + n_base + 16 + (lane & 15)] = (_Float16)acc1[r];
    }
  } else {
    // transposed store: 8 consecutive M per lane -> contiguous along s.
    const int b      = blockIdx.x >> 5;        // 64-row tile lies in one batch
    const int s_base = m_base - b * 2048 + roff;
    PackU p0, p1;
    #pragma unroll
    for (int r = 0; r < 8; ++r) { p0.h[r] = (_Float16)acc0[r]; p1.h[r] = (_Float16)acc1[r]; }
    const int u0 = n_base + (lane & 15);
    *reinterpret_cast<uint4*>(VT + ((size_t)b * 768 + u0) * 2048 + s_base)      = p0.q;
    *reinterpret_cast<uint4*>(VT + ((size_t)b * 768 + u0 + 16) * 2048 + s_base) = p1.q;
  }
}

// ---------------------------------------------------------------- flash attention
// grid (64, 4), block 256, dynamic LDS 160128 B.
// LDS: Qs 32x768 f16 | Os 32x768 f32 | Ssc 32x64 f32 | Pp 32x64 f16 | m/l/a rows
__global__ __launch_bounds__(256) void k_attn(const _Float16* __restrict__ Qh,
                                              const _Float16* __restrict__ Kh,
                                              const _Float16* __restrict__ VT,
                                              float* __restrict__ out) {
  extern __shared__ __align__(16) char smem[];
  _Float16* Qs   = reinterpret_cast<_Float16*>(smem);                       // 49152 B
  float*    Os   = reinterpret_cast<float*>(smem + 49152);                  // 98304 B
  float*    Ssc  = reinterpret_cast<float*>(smem + 147456);                 // 8192 B
  _Float16* Pp   = reinterpret_cast<_Float16*>(smem + 155648);              // 4096 B
  float*    mrow = reinterpret_cast<float*>(smem + 159744);
  float*    lrow = reinterpret_cast<float*>(smem + 159744 + 128);
  float*    arow = reinterpret_cast<float*>(smem + 159744 + 256);

  const int tid  = threadIdx.x;
  const int lane = tid & 31;
  const int w    = tid >> 5;
  const int b    = blockIdx.y;
  const int q0   = blockIdx.x * 32;
  const float scale = 0.03608439182435161f;    // 1/sqrt(768)

  { // stage Q tile (contiguous 48 KB)
    const uint4* src = reinterpret_cast<const uint4*>(Qh + ((size_t)b * 2048 + q0) * 768);
    uint4* dst = reinterpret_cast<uint4*>(Qs);
    for (int i = tid; i < 3072; i += 256) dst[i] = src[i];
  }
  for (int i = tid; i < 32 * 768; i += 256) Os[i] = 0.0f;
  if (tid < 32) { mrow[tid] = -1e30f; lrow[tid] = 0.0f; }
  __syncthreads();

  const int qg   = w >> 2;                     // 0..1 : 16-row q group
  const int sg   = w & 3;                      // 0..3 : 16-col s group (scores)
  const int us   = w & 3;                      // 0..3 : 192-col u strip (PV)
  const int roff = (lane & 16) ? 8 : 0;

  for (int s0 = 0; s0 < 2048; s0 += 64) {
    // ---- scores: S = scale * Q K^T (16x16 tile per wave, K=768)
    const _Float16* qrow = Qs + (size_t)(qg * 16 + (lane & 15)) * 768;
    const _Float16* krow = Kh + ((size_t)b * 2048 + s0 + sg * 16 + (lane & 15)) * 768;
    if (s0 + 64 < 2048)
      __builtin_prefetch(krow + 64 * 768, 0, 1);   // next chunk's K rows
    v8f se = {}, so = {};
    for (int k0 = 0; k0 < 768; k0 += 64) {         // two independent WMMA chains
      se = WMMA16x16x32(load_a16(qrow + k0,      lane), load_b16(krow + k0,      lane), se);
      so = WMMA16x16x32(load_a16(qrow + k0 + 32, lane), load_b16(krow + k0 + 32, lane), so);
    }
    #pragma unroll
    for (int r = 0; r < 8; ++r)
      Ssc[(qg * 16 + roff + r) * 64 + sg * 16 + (lane & 15)] = (se[r] + so[r]) * scale;
    __syncthreads();

    // ---- online softmax: one thread per q row
    if (tid < 32) {
      const float* srow = Ssc + tid * 64;
      float mx = mrow[tid];
      for (int j = 0; j < 64; ++j) mx = fmaxf(mx, srow[j]);
      const float alpha = __expf(mrow[tid] - mx);
      float sum = 0.0f;
      _Float16* prow = Pp + tid * 64;
      for (int j = 0; j < 64; ++j) {
        const float p = __expf(srow[j] - mx);
        sum += p;
        prow[j] = (_Float16)p;
      }
      lrow[tid] = lrow[tid] * alpha + sum;
      mrow[tid] = mx;
      arow[tid] = alpha;
    }
    __syncthreads();

    // ---- PV: O = alpha*O + P V   (wave = q group x 192-wide u strip)
    const v16h a0 = load_a16(Pp + (size_t)(qg * 16 + (lane & 15)) * 64 + 0,  lane);
    const v16h a1 = load_a16(Pp + (size_t)(qg * 16 + (lane & 15)) * 64 + 32, lane);
    float alph[8];
    #pragma unroll
    for (int r = 0; r < 8; ++r) alph[r] = arow[qg * 16 + roff + r];
    const _Float16* vbase = VT + (size_t)b * 768 * 2048 + s0;
    for (int t = 0; t < 12; ++t) {
      const int ucol = us * 192 + t * 16 + (lane & 15);
      const _Float16* vrow = vbase + (size_t)ucol * 2048;
      const v16h b0 = load_b16(vrow + 0,  lane);
      const v16h b1 = load_b16(vrow + 32, lane);
      v8f c;
      #pragma unroll
      for (int r = 0; r < 8; ++r)
        c[r] = Os[(qg * 16 + roff + r) * 768 + ucol] * alph[r];
      c = WMMA16x16x32(a0, b0, c);
      c = WMMA16x16x32(a1, b1, c);
      #pragma unroll
      for (int r = 0; r < 8; ++r)
        Os[(qg * 16 + roff + r) * 768 + ucol] = c[r];
    }
    // score(i+1)->softmax(i+1) barrier also orders PV(i) before Pp/arow reuse
  }

  __syncthreads();
  if (tid < 32) arow[tid] = 1.0f / lrow[tid];
  __syncthreads();
  float* outp = out + ((size_t)b * 2048 + q0) * 768;
  for (int i = tid; i < 32 * 768; i += 256)
    outp[i] = Os[i] * arow[i / 768];
}

// ---------------------------------------------------------------- launch
extern "C" void kernel_launch(void* const* d_in, const int* in_sizes, int n_in,
                              void* d_out, int out_size, void* d_ws, size_t ws_size,
                              hipStream_t stream) {
  (void)in_sizes; (void)n_in; (void)out_size; (void)ws_size;
  const float* x  = (const float*)d_in[0];
  const float* Wq = (const float*)d_in[1];
  const float* Wk = (const float*)d_in[2];
  const float* Wv = (const float*)d_in[3];
  float* out = (float*)d_out;

  char* ws = (char*)d_ws;
  size_t off = 0;
  auto alloc = [&](size_t bytes) -> char* {
    char* p = ws + off;
    off = (off + bytes + 255) & ~(size_t)255;
    return p;
  };
  _Float16* xh = (_Float16*)alloc((size_t)8192 * 768 * 2);      // x in f16
  _Float16* Wt = (_Float16*)alloc((size_t)3 * 768 * 768 * 2);   // W^T f16 [3][U][D]
  _Float16* Qh = (_Float16*)alloc((size_t)8192 * 768 * 2);      // Q row-major
  _Float16* Kh = (_Float16*)alloc((size_t)8192 * 768 * 2);      // K row-major
  _Float16* VT = (_Float16*)alloc((size_t)4 * 768 * 2048 * 2);  // V^T [B][U][S]
  // total ws use ~53.9 MB

  k_cvt_x<<<dim3(6144), dim3(256), 0, stream>>>(x, xh, 8192 * 768 / 4);
  k_wt   <<<dim3(3, 768, 3), dim3(256), 0, stream>>>(Wq, Wk, Wv, Wt);
  k_qkv  <<<dim3(128, 12, 3), dim3(256), 0, stream>>>(xh, Wt, Qh, Kh, VT);
  k_attn <<<dim3(64, 4), dim3(256), 160128, stream>>>(Qh, Kh, VT, out);
}
